// RPN_40321152975414
// MI455X (gfx1250) — matrix operand order repaired and checked by reading, not compiled, not run on hardware
//
#include <hip/hip_runtime.h>
#include <hip/hip_bf16.h>

typedef __attribute__((ext_vector_type(16))) _Float16 v16h;
typedef __attribute__((ext_vector_type(8)))  _Float16 v8h;
typedef __attribute__((ext_vector_type(8)))  float    v8f;
typedef unsigned int u32x4 __attribute__((ext_vector_type(4)));
typedef int          i32x4 __attribute__((ext_vector_type(4)));
typedef int          i32x8 __attribute__((ext_vector_type(8)));

#define BB      4
#define CC      256
#define HH      100
#define WW      100
#define AA      9
#define NPIX    40000      /* B*H*W  */
#define NANCH   90000      /* H*W*A per image */
#define PADW    102
#define KPRE    2000
#define KPOST   1000
#define NMS_T   0.7f
#define IMGSZ   800.0f
#define MINSZ   1e-3f
#define BCLIP   4.1351666f /* log(1000/16) */

/* workspace byte offsets (all 256B aligned) */
#define OFF_FEAT 0u
#define OFF_WB   21307392u
#define OFF_HW   22487040u
#define OFF_T    22511616u
#define OFF_HOUT 42991616u
#define OFF_BOX  50671616u
#define OFF_SCR  56431616u
#define OFF_TB   57871616u
#define OFF_TS   57999616u

/* ---- Tensor Data Mover availability / arity detection ---- */
#if defined(__has_builtin)
#if __has_builtin(__builtin_amdgcn_tensor_load_to_lds)
#define HAVE_TDM 1
#endif
#endif
#if __has_include(<hip/amd_detail/amd_gfx1250_TDM.h>)
#define TDM_6ARG 1
#endif

#if defined(HAVE_TDM)
/* 2-D tile DMA: tileRows rows of 32 f16 (64B) each, global row stride 256 f16,
 * packed contiguously into LDS at ldsByteOff. */
__device__ __forceinline__ void tdm_load_wtile(const _Float16* g, unsigned ldsByteOff) {
    unsigned long long ga = (unsigned long long)(size_t)g;
    u32x4 g0 = { 1u,                                   /* count=1, user desc   */
                 ldsByteOff,                           /* lds_addr             */
                 (unsigned)(ga & 0xffffffffu),         /* global_addr[31:0]    */
                 (unsigned)((ga >> 32) & 0x1ffffffu) | (2u << 30) }; /* +type=2 */
    i32x8 g1 = { (int)0x00010000,                      /* data_size=2B         */
                 0,                                    /* atomic barrier off   */
                 0x10,                                 /* tensor_dim0=1<<20    */
                 (32 << 16) | 0x10,                    /* tile_dim0=32, td1 hi */
                 256,                                  /* tile_dim1=256 rows   */
                 256,                                  /* dim0_stride=256 elem */
                 0, 0 };
    i32x4 z4 = { 0, 0, 0, 0 };
#if defined(TDM_6ARG)
    i32x8 z8 = { 0, 0, 0, 0, 0, 0, 0, 0 };
    __builtin_amdgcn_tensor_load_to_lds(g0, g1, z4, z4, z8, 0);
#else
    __builtin_amdgcn_tensor_load_to_lds(g0, g1, z4, z4, 0);
#endif
}
#endif

/* ---------------- prep: fp32 -> f16 relayouts ---------------- */

__global__ __launch_bounds__(256)
void prep_feat(const float* __restrict__ feat, _Float16* __restrict__ out) {
    int idx = blockIdx.x * 256 + threadIdx.x;
    const int total = BB * PADW * PADW * CC;
    if (idx >= total) return;
    int c  = idx & 255;
    int t  = idx >> 8;
    int xx = t % PADW; t /= PADW;
    int yy = t % PADW;
    int b  = t / PADW;
    float v = 0.0f;
    if (yy >= 1 && yy <= HH && xx >= 1 && xx <= WW)
        v = feat[(((size_t)b * CC + c) * HH + (yy - 1)) * WW + (xx - 1)];
    out[idx] = (_Float16)v;
}

__global__ __launch_bounds__(256)
void prep_wb(const float* __restrict__ cw, _Float16* __restrict__ out) {
    int idx = blockIdx.x * 256 + threadIdx.x;
    const int total = 9 * 256 * 256;
    if (idx >= total) return;
    int k = idx & 255;
    int t = idx >> 8;
    int n = t & 255;
    int tap = t >> 8;
    int ky = tap / 3, kx = tap % 3;
    out[idx] = (_Float16)cw[(((size_t)n * 256 + k) * 3 + ky) * 3 + kx];
}

__global__ __launch_bounds__(256)
void prep_hw(const float* __restrict__ clsw, const float* __restrict__ bbw,
             _Float16* __restrict__ out) {
    int idx = blockIdx.x * 256 + threadIdx.x;
    const int total = 48 * 256;
    if (idx >= total) return;
    int c = idx & 255;
    int n = idx >> 8;
    float v = 0.0f;
    if (n < 9)       v = clsw[(size_t)n * 256 + c];
    else if (n < 45) v = bbw[(size_t)(n - 9) * 256 + c];
    out[idx] = (_Float16)v;
}

/* ---------------- conv 3x3 as implicit GEMM, WMMA f16 ----------------
 * M = NPIX (block: 128 rows = 8 waves x 16), N = 256 (16 n-tiles/wave),
 * K = 9 taps x 256 ch, chunked by 32.  B tile staged in LDS via TDM
 * (double-buffered, wave0 drives the DMA + s_wait_tensorcnt). */

__global__ __launch_bounds__(256)
void conv3x3_wmma(const _Float16* __restrict__ feat,  /* padded NHWC f16 */
                  const _Float16* __restrict__ wB,    /* [9][256n][256k] */
                  const float*    __restrict__ bias,
                  _Float16*       __restrict__ tbuf)  /* [NPIX][256]     */
{
    __shared__ uint4 Bt4[2][1024];                    /* 2 x 16KB B tiles */

    const int tid  = threadIdx.x;
    const int lane = tid & 31;
    const int wv   = tid >> 5;
    const int rowBase = blockIdx.x * 128 + wv * 16;

    /* A fragment: lane<16 -> K 0..15, lane>=16 -> K 16..31 */
    const int koff = (lane >> 4) ? 16 : 0;
    int p  = rowBase + (lane & 15);
    int pc = p < NPIX ? p : NPIX - 1;
    int bimg = pc / (HH * WW);
    int rem  = pc % (HH * WW);
    int y = rem / WW, x = rem % WW;

    /* B fragment: lane<16 -> K 0..15 of col n, lane>=16 -> K 16..31 */
    const int nrow = lane & 15;
    const int kf   = (lane >> 4) ? 16 : 0;

    v8f acc[16];
    #pragma unroll
    for (int i = 0; i < 16; ++i)
        #pragma unroll
        for (int j = 0; j < 8; ++j) acc[i][j] = 0.0f;

#if defined(HAVE_TDM)
    const unsigned ldsOff0 = (unsigned)(size_t)(void*)&Bt4[0][0];
    const unsigned ldsOff1 = (unsigned)(size_t)(void*)&Bt4[1][0];
    if (wv == 0) tdm_load_wtile(wB, ldsOff0);         /* prologue: chunk 0 */
#endif

    for (int ck = 0; ck < 72; ++ck) {
        const int t  = ck >> 3;
        const int c0 = (ck & 7) * 32;
        const int ky = t / 3, kx = t % 3;
        const _Float16* arow =
            feat + (size_t)((bimg * PADW + y + ky) * PADW + (x + kx)) * CC;

#if defined(HAVE_TDM)
        if (wv == 0) {
            if (ck + 1 < 72) {
                const int t2 = (ck + 1) >> 3;
                const int c2 = ((ck + 1) & 7) * 32;
                tdm_load_wtile(wB + (size_t)t2 * 65536 + c2,
                               ((ck + 1) & 1) ? ldsOff1 : ldsOff0);
                __builtin_amdgcn_s_wait_tensorcnt((short)1); /* chunk ck done */
            } else {
                __builtin_amdgcn_s_wait_tensorcnt((short)0);
            }
        }
        __syncthreads();                       /* publish buffer ck&1        */
        _Float16* Bt = (_Float16*)Bt4[ck & 1];
#else
        /* fallback: cooperative staging, thread tid owns row n=tid (64B)   */
        {
            const uint4* src =
                (const uint4*)(wB + ((size_t)(t * 256 + tid)) * 256 + c0);
            #pragma unroll
            for (int i = 0; i < 4; ++i) Bt4[0][tid * 4 + i] = src[i];
        }
        __syncthreads();
        _Float16* Bt = (_Float16*)Bt4[0];
#endif

        /* A fragment from global (two 16B loads / lane) */
        v8h a0 = *(const v8h*)(arow + c0 + koff);
        v8h a1 = *(const v8h*)(arow + c0 + koff + 8);
        v16h afrag;
        #pragma unroll
        for (int i = 0; i < 8; ++i) { afrag[i] = a0[i]; afrag[i + 8] = a1[i]; }
        __builtin_prefetch(arow + c0 + 32, 0, 3);   /* next K-chunk of A */

        #pragma unroll
        for (int nt = 0; nt < 16; ++nt) {
            const v8h* bp = (const v8h*)&Bt[(nt * 16 + nrow) * 32 + kf];
            v8h b0 = bp[0], b1 = bp[1];
            v16h bfrag;
            #pragma unroll
            for (int i = 0; i < 8; ++i) { bfrag[i] = b0[i]; bfrag[i + 8] = b1[i]; }
            acc[nt] = __builtin_amdgcn_wmma_f32_16x16x32_f16(
                false, afrag, false, bfrag, (short)0, acc[nt], false, false);
        }
        __syncthreads();                       /* buffer reuse guard */
    }

    /* epilogue: bias + ReLU -> f16.  C/D: lane<16 M=r, N=lane;
     * lane>=16 M=8+r, N=lane-16 */
    const int mhi = (lane >> 4) * 8;
    #pragma unroll
    for (int nt = 0; nt < 16; ++nt) {
        const int n = nt * 16 + nrow;
        const float bn = bias[n];
        #pragma unroll
        for (int r = 0; r < 8; ++r) {
            const int pp = rowBase + mhi + r;
            if (pp < NPIX) {
                float v = acc[nt][r] + bn;
                v = v > 0.0f ? v : 0.0f;
                tbuf[(size_t)pp * 256 + n] = (_Float16)v;
            }
        }
    }
}

/* ---------------- 1x1 heads: M=NPIX, K=256, N=48 WMMA GEMM ---------------- */

__global__ __launch_bounds__(256)
void head_wmma(const _Float16* __restrict__ tbuf,
               const _Float16* __restrict__ hwgt,   /* [48][256] f16 */
               float* __restrict__ hout)            /* [NPIX][48] f32 */
{
    const int tid  = threadIdx.x;
    const int lane = tid & 31;
    const int wv   = tid >> 5;
    const int rowBase = blockIdx.x * 128 + wv * 16;
    const int koff = (lane >> 4) ? 16 : 0;
    const int kf   = (lane >> 4) ? 16 : 0;
    const int nrow = lane & 15;

    int p  = rowBase + (lane & 15);
    int pc = p < NPIX ? p : NPIX - 1;
    const _Float16* arow = tbuf + (size_t)pc * 256;

    v8f acc[3];
    #pragma unroll
    for (int i = 0; i < 3; ++i)
        #pragma unroll
        for (int j = 0; j < 8; ++j) acc[i][j] = 0.0f;

    for (int kc = 0; kc < 8; ++kc) {
        const int c0 = kc * 32;
        v8h a0 = *(const v8h*)(arow + c0 + koff);
        v8h a1 = *(const v8h*)(arow + c0 + koff + 8);
        v16h afrag;
        #pragma unroll
        for (int i = 0; i < 8; ++i) { afrag[i] = a0[i]; afrag[i + 8] = a1[i]; }
        #pragma unroll
        for (int nt = 0; nt < 3; ++nt) {
            const v8h* bp =
                (const v8h*)(hwgt + (size_t)(nt * 16 + nrow) * 256 + c0 + kf);
            v8h b0 = bp[0], b1 = bp[1];
            v16h bfrag;
            #pragma unroll
            for (int i = 0; i < 8; ++i) { bfrag[i] = b0[i]; bfrag[i + 8] = b1[i]; }
            acc[nt] = __builtin_amdgcn_wmma_f32_16x16x32_f16(
                false, afrag, false, bfrag, (short)0, acc[nt], false, false);
        }
    }

    const int mhi = (lane >> 4) * 8;
    #pragma unroll
    for (int nt = 0; nt < 3; ++nt) {
        const int n = nt * 16 + nrow;
        #pragma unroll
        for (int r = 0; r < 8; ++r) {
            const int pp = rowBase + mhi + r;
            if (pp < NPIX) hout[(size_t)pp * 48 + n] = acc[nt][r];
        }
    }
}

/* ---------------- anchor decode ---------------- */

__global__ __launch_bounds__(256)
void decode_kernel(const float* __restrict__ hout,
                   const float* __restrict__ clsb, const float* __restrict__ bbb,
                   float* __restrict__ boxes, float* __restrict__ scores)
{
    int idx = blockIdx.x * 256 + threadIdx.x;
    const int total = BB * NANCH;
    if (idx >= total) return;
    const int b   = idx / NANCH;
    const int rem = idx % NANCH;      /* = hw*9 + a, matches [H,W,A] reshape */
    const int hw  = rem / AA;
    const int a   = rem % AA;
    const int p   = b * (HH * WW) + hw;

    const float scale = 32.0f * (float)(1 << (a % 3));
    const float ratio = 0.5f  * (float)(1 << (a / 3));
    const float hr = sqrtf(ratio);
    const float wr = 1.0f / hr;
    const float wa = wr * scale, ha = hr * scale;
    const float cxa = (float)(hw % WW) * 8.0f;
    const float cya = (float)(hw / WW) * 8.0f;

    const float* row = hout + (size_t)p * 48;
    const float logit = row[a] + clsb[a];
    const float dx = row[9 + a * 4 + 0] + bbb[a * 4 + 0];
    const float dy = row[9 + a * 4 + 1] + bbb[a * 4 + 1];
    float dw = row[9 + a * 4 + 2] + bbb[a * 4 + 2];
    float dh = row[9 + a * 4 + 3] + bbb[a * 4 + 3];
    dw = fminf(dw, BCLIP);
    dh = fminf(dh, BCLIP);
    const float cx = dx * wa + cxa;
    const float cy = dy * ha + cya;
    const float w  = __expf(dw) * wa;
    const float h  = __expf(dh) * ha;

    float* bx = boxes + (size_t)idx * 4;
    bx[0] = cx - 0.5f * w;
    bx[1] = cy - 0.5f * h;
    bx[2] = cx + 0.5f * w;
    bx[3] = cy + 0.5f * h;
    scores[idx] = logit;
}

/* ---------------- per-image top-2000: value bisection + ordered compaction */

__global__ __launch_bounds__(1024)
void topk_select(const float* __restrict__ scores, const float* __restrict__ boxes,
                 float* __restrict__ ts, float* __restrict__ tb)
{
    const int b   = blockIdx.x;
    const int tid = threadIdx.x;
    const float* s  = scores + (size_t)b * NANCH;
    const float* bx = boxes  + (size_t)b * NANCH * 4;
    __shared__ int cnt[1024];
    __shared__ int pref[1024];
    __shared__ int totS;

    const int CH = (NANCH + 1023) / 1024;    /* 88 */
    const int i0 = tid * CH;
    const int i1 = (i0 + CH) < NANCH ? (i0 + CH) : NANCH;

    float lo = -1e4f, hi = 1e4f;
    for (int it = 0; it < 48; ++it) {
        const float mid = 0.5f * (lo + hi);
        int c = 0;
        for (int i = i0; i < i1; ++i) c += (s[i] > mid) ? 1 : 0;
        cnt[tid] = c;
        __syncthreads();
        if (tid == 0) {
            int tot = 0;
            for (int j = 0; j < 1024; ++j) tot += cnt[j];
            totS = tot;
        }
        __syncthreads();
        if (totS >= KPRE) lo = mid; else hi = mid;
        __syncthreads();
    }
    const float thr = lo;

    for (int i = tid; i < KPRE; i += 1024) {
        ts[(size_t)b * KPRE + i] = -1.0f;
        float* o = tb + ((size_t)b * KPRE + i) * 4;
        o[0] = 0.0f; o[1] = 0.0f; o[2] = 0.0f; o[3] = 0.0f;
    }
    __syncthreads();

    int c = 0;
    for (int i = i0; i < i1; ++i) c += (s[i] > thr) ? 1 : 0;
    cnt[tid] = c;
    __syncthreads();
    if (tid == 0) {
        int run = 0;
        for (int j = 0; j < 1024; ++j) { pref[j] = run; run += cnt[j]; }
    }
    __syncthreads();

    int pos = pref[tid];
    for (int i = i0; i < i1; ++i) {
        const float sv = s[i];
        if (sv > thr) {
            if (pos < KPRE) {
                float x1 = fminf(fmaxf(bx[(size_t)i * 4 + 0], 0.0f), IMGSZ);
                float y1 = fminf(fmaxf(bx[(size_t)i * 4 + 1], 0.0f), IMGSZ);
                float x2 = fminf(fmaxf(bx[(size_t)i * 4 + 2], 0.0f), IMGSZ);
                float y2 = fminf(fmaxf(bx[(size_t)i * 4 + 3], 0.0f), IMGSZ);
                const bool valid = ((x2 - x1) >= MINSZ) && ((y2 - y1) >= MINSZ);
                const float pr = valid ? 1.0f / (1.0f + __expf(-sv)) : -1.0f;
                float* o = tb + ((size_t)b * KPRE + pos) * 4;
                o[0] = x1; o[1] = y1; o[2] = x2; o[3] = y2;
                ts[(size_t)b * KPRE + pos] = pr;
            }
            ++pos;
        }
    }
}

/* ---------------- greedy NMS, 1000 outputs/image ---------------- */

__global__ __launch_bounds__(256)
void nms_kernel(const float* __restrict__ ts, const float* __restrict__ tb,
                float* __restrict__ out)
{
    const int b   = blockIdx.x;
    const int tid = threadIdx.x;
    __shared__ float  sb[KPRE];
    __shared__ float4 bbx[KPRE];
    __shared__ float  rv[256];
    __shared__ int    ri[256];
    __shared__ float4 wbox;

    for (int i = tid; i < KPRE; i += 256) {
        sb[i] = ts[(size_t)b * KPRE + i];
        const float* o = tb + ((size_t)b * KPRE + i) * 4;
        bbx[i] = make_float4(o[0], o[1], o[2], o[3]);
    }
    __syncthreads();

    for (int it = 0; it < KPOST; ++it) {
        float bv = -3.4e38f; int bi = 0;
        for (int i = tid; i < KPRE; i += 256) {
            const float v = sb[i];
            if (v > bv) { bv = v; bi = i; }
        }
        rv[tid] = bv; ri[tid] = bi;
        __syncthreads();
        for (int s2 = 128; s2 > 0; s2 >>= 1) {
            if (tid < s2) {
                const float v2 = rv[tid + s2]; const int j2 = ri[tid + s2];
                if (v2 > rv[tid] || (v2 == rv[tid] && j2 < ri[tid])) {
                    rv[tid] = v2; ri[tid] = j2;
                }
            }
            __syncthreads();
        }
        if (tid == 0) {
            const int wi = ri[0];
            const float sc = rv[0];
            const float4 wb = bbx[wi];
            wbox = wb;
            float* o = out + ((size_t)b * KPOST + it) * 5;
            const bool keep = sc > 0.0f;
            o[0] = keep ? wb.x : 0.0f;
            o[1] = keep ? wb.y : 0.0f;
            o[2] = keep ? wb.z : 0.0f;
            o[3] = keep ? wb.w : 0.0f;
            o[4] = keep ? sc   : 0.0f;
        }
        __syncthreads();
        const float4 wb = wbox;
        const float a1 = (wb.z - wb.x) * (wb.w - wb.y);
        for (int i = tid; i < KPRE; i += 256) {
            const float4 cb = bbx[i];
            const float xx1 = fmaxf(wb.x, cb.x), yy1 = fmaxf(wb.y, cb.y);
            const float xx2 = fminf(wb.z, cb.z), yy2 = fminf(wb.w, cb.w);
            const float inter = fmaxf(xx2 - xx1, 0.0f) * fmaxf(yy2 - yy1, 0.0f);
            const float a2 = (cb.z - cb.x) * (cb.w - cb.y);
            const float iou = inter / (a1 + a2 - inter + 1e-9f);
            if (iou >= NMS_T) sb[i] = -1.0f;
        }
        __syncthreads();
    }
}

/* ---------------- launch ---------------- */

extern "C" void kernel_launch(void* const* d_in, const int* in_sizes, int n_in,
                              void* d_out, int out_size, void* d_ws, size_t ws_size,
                              hipStream_t stream) {
    const float* feat   = (const float*)d_in[0];
    const float* conv_w = (const float*)d_in[1];
    const float* conv_b = (const float*)d_in[2];
    const float* cls_w  = (const float*)d_in[3];
    const float* cls_b  = (const float*)d_in[4];
    const float* bbox_w = (const float*)d_in[5];
    const float* bbox_b = (const float*)d_in[6];

    char* ws = (char*)d_ws;
    _Float16* featf16 = (_Float16*)(ws + OFF_FEAT);
    _Float16* wB      = (_Float16*)(ws + OFF_WB);
    _Float16* hwgt    = (_Float16*)(ws + OFF_HW);
    _Float16* tbuf    = (_Float16*)(ws + OFF_T);
    float*    hout    = (float*)(ws + OFF_HOUT);
    float*    boxes   = (float*)(ws + OFF_BOX);
    float*    scores  = (float*)(ws + OFF_SCR);
    float*    tbx     = (float*)(ws + OFF_TB);
    float*    tsc     = (float*)(ws + OFF_TS);
    float*    out     = (float*)d_out;

    {
        const int total = BB * PADW * PADW * CC;
        prep_feat<<<(total + 255) / 256, 256, 0, stream>>>(feat, featf16);
    }
    {
        const int total = 9 * 256 * 256;
        prep_wb<<<(total + 255) / 256, 256, 0, stream>>>(conv_w, wB);
    }
    {
        const int total = 48 * 256;
        prep_hw<<<(total + 255) / 256, 256, 0, stream>>>(cls_w, bbox_w, hwgt);
    }
    conv3x3_wmma<<<(NPIX + 127) / 128, 256, 0, stream>>>(featf16, wB, conv_b, tbuf);
    head_wmma<<<(NPIX + 127) / 128, 256, 0, stream>>>(tbuf, hwgt, hout);
    {
        const int total = BB * NANCH;
        decode_kernel<<<(total + 255) / 256, 256, 0, stream>>>(hout, cls_b, bbox_b,
                                                               boxes, scores);
    }
    topk_select<<<BB, 1024, 0, stream>>>(scores, boxes, tsc, tbx);
    nms_kernel<<<BB, 256, 0, stream>>>(tsc, tbx, out);
}
